// GatedPooling_89404039234016
// MI455X (gfx1250) — matrix operand order repaired and checked by reading, not compiled, not run on hardware
//
#include <hip/hip_runtime.h>
#include <hip/hip_bf16.h>
#include <stdint.h>

// ---------------------------------------------------------------------------
// CDNA5 (gfx1250) fused gated-pooling:
//   gates = sigmoid(LN(x@Wg^T + bg)); feats = gelu(LN(x@Wf^T + bf))
//   pooled = scatter_mean(gates*feats, cluster); out = pooled[cluster]
// GEMMs run on v_wmma_f32_16x16x32_f16 (f32 accumulate).
// ---------------------------------------------------------------------------

typedef __attribute__((ext_vector_type(16))) _Float16 v16h;
typedef __attribute__((ext_vector_type(8)))  float    v8f;
typedef __attribute__((ext_vector_type(8)))  uint32_t v8u;

#define HIDDEN   512
#define NCLUST   2048
#define BM       32                 // rows per block (2 WMMA M-tiles)
#define NT_TILES (HIDDEN / 16)      // 32 output-column tiles
#define KT_TILES (HIDDEN / 32)      // 16 K-steps of 32

__device__ __forceinline__ uint32_t pack_h2(float a, float b) {
  _Float16 ha = (_Float16)a, hb = (_Float16)b;
  uint16_t ua = __builtin_bit_cast(uint16_t, ha);
  uint16_t ub = __builtin_bit_cast(uint16_t, hb);
  return (uint32_t)ua | ((uint32_t)ub << 16);
}

// ---- zero the cluster accumulators (ws is poisoned; must re-zero each call)
__global__ void k_zero(float* __restrict__ p, int n) {
  int i = blockIdx.x * blockDim.x + threadIdx.x;
  if (i < n) p[i] = 0.0f;
}

// ---- convert both weight matrices fp32 -> f16, pre-swizzled into the WMMA
//      B-fragment layout in contiguous k order (not the ISA table order), 16-bit B 32x16:
//      lane = khalf*16 + n_local ; VGPR j holds K = khalf*16 + 2j, 2j+1
//      bfrag dword index = (((widx*32 + nt)*16 + kt)*32 + lane)*8 + j
__global__ void k_swizzle_w(const float* __restrict__ gate_w,
                            const float* __restrict__ feat_w,
                            uint32_t* __restrict__ bfrag) {
  int idx = blockIdx.x * blockDim.x + threadIdx.x;   // 0 .. 262143
  int j    =  idx        & 7;
  int lane = (idx >> 3)  & 31;
  int kt   = (idx >> 8)  & 15;
  int nt   = (idx >> 12) & 31;
  int widx =  idx >> 17;
  const float* W = widx ? feat_w : gate_w;           // [out][in], row-major
  int n_local = lane & 15;
  int khalf   = lane >> 4;
  int k = kt * 32 + khalf * 16 + 2 * j;              // B[k][o] = W[o][k]
  int o = nt * 16 + n_local;
  const float* src = W + (size_t)o * HIDDEN + k;
  bfrag[idx] = pack_h2(src[0], src[1]);
}

__device__ __forceinline__ float gated_val(float gv, float fv,
                                           float mu_g, float rs_g, float gg, float gb,
                                           float mu_f, float rs_f, float fg, float fb) {
  float gn = (gv - mu_g) * rs_g * gg + gb;
  float s  = 1.0f / (1.0f + expf(-gn));
  float fn = (fv - mu_f) * rs_f * fg + fb;
  float e  = 0.5f * fn * (1.0f + erff(fn * 0.70710678118654752f));  // exact GELU
  return s * e;
}

// ---- fused: dual WMMA GEMM + bias + LN + sigmoid/gelu + product + scatter-add
__global__ void __launch_bounds__(256)
k_fused(const float* __restrict__ x,
        const int*   __restrict__ cluster,
        const uint32_t* __restrict__ bfrag,
        const float* __restrict__ gate_b, const float* __restrict__ gate_lng,
        const float* __restrict__ gate_lnb,
        const float* __restrict__ feat_b, const float* __restrict__ feat_lng,
        const float* __restrict__ feat_lnb,
        float* __restrict__ sums, float* __restrict__ counts, int n_nodes) {
  extern __shared__ char smem[];
  uint32_t* s_afrag = (uint32_t*)smem;                 //  32 KB: A frags [mt][kt][lane][j]
  float*    s_out   = (float*)(smem + 32768);          // 128 KB: [branch][row][col]
  float*    s_x     = s_out;                           //  alias: fp32 x staging (64 KB)

  const int t    = threadIdx.x;
  const int lane = t & 31;
  const int wave = t >> 5;
  const int row0 = blockIdx.x * BM;

  // --- stage 1: coalesced fp32 x tile -> LDS
  {
    const float4* xg = (const float4*)(x + (size_t)row0 * HIDDEN);
    float4* s4 = (float4*)s_x;
    const float4 z4 = make_float4(0.f, 0.f, 0.f, 0.f);
    for (int i = t; i < BM * HIDDEN / 4; i += 256) {
      int r = i >> 7;                                  // 128 float4 per row
      s4[i] = (row0 + r < n_nodes) ? xg[i] : z4;
    }
  }
  __syncthreads();

  // --- stage 2: build f16 A fragments (16-bit A 16x32, contiguous k order as for B)
  for (int idx = t; idx < 2 * KT_TILES * 32 * 8; idx += 256) {
    int j  =  idx        & 7;
    int l  = (idx >> 3)  & 31;
    int kt = (idx >> 8)  & 15;
    int mt =  idx >> 12;
    int m     = mt * 16 + (l & 15);
    int khalf = l >> 4;
    int kl = khalf * 16 + 2 * j;
    int k  = kt * 32 + kl;
    s_afrag[idx] = pack_h2(s_x[m * HIDDEN + k], s_x[m * HIDDEN + k + 1]);
  }
  __syncthreads();

  // --- stage 3: dual GEMM. 64 jobs = {gate,feat} x 32 N-tiles; 8 per wave.
  const int khalf = lane >> 4;
  const int nl    = lane & 15;
  for (int job = 0; job < 8; ++job) {
    int jobid = wave * 8 + job;
    int widx  = jobid >> 5;
    int nt    = jobid & 31;
    v8f acc0 = {};                                     // M-tile 0 (rows 0..15)
    v8f acc1 = {};                                     // M-tile 1 (rows 16..31)
    const v8u* bq = (const v8u*)bfrag + ((size_t)(widx * 32 + nt) * KT_TILES) * 32 + lane;
    const v8u* a0 = (const v8u*)s_afrag + lane;
    const v8u* a1 = (const v8u*)s_afrag + KT_TILES * 32 + lane;
#pragma unroll
    for (int kt = 0; kt < KT_TILES; ++kt) {
      v16h bf  = __builtin_bit_cast(v16h, bq[kt * 32]);
      v16h af0 = __builtin_bit_cast(v16h, a0[kt * 32]);
      v16h af1 = __builtin_bit_cast(v16h, a1[kt * 32]);
      acc0 = __builtin_amdgcn_wmma_f32_16x16x32_f16(false, af0, false, bf,
                                                    (short)0, acc0, false, false);
      acc1 = __builtin_amdgcn_wmma_f32_16x16x32_f16(false, af1, false, bf,
                                                    (short)0, acc1, false, false);
    }
    // C layout: VGPR r, lanes 0-15 -> M=r, lanes 16-31 -> M=r+8, N=lane&15
    float* og = s_out + (size_t)widx * BM * HIDDEN;
    int col = nt * 16 + nl;
#pragma unroll
    for (int r = 0; r < 8; ++r) {
      int m = r + 8 * khalf;
      og[m * HIDDEN + col]        = acc0[r];
      og[(m + 16) * HIDDEN + col] = acc1[r];
    }
  }
  __syncthreads();

  // --- stage 4: bias + LayerNorm + activations + product + scatter-mean acc.
  {
    int row = t >> 3;                                  // 0..31, 8 threads/row
    int sub = t & 7;
    const float4* og4 = (const float4*)(s_out + (size_t)row * HIDDEN);
    const float4* of4 = (const float4*)(s_out + (size_t)(BM + row) * HIDDEN);
    const float4* gb4  = (const float4*)gate_b;
    const float4* fb4  = (const float4*)feat_b;
    const float4* gg4  = (const float4*)gate_lng;
    const float4* gbe4 = (const float4*)gate_lnb;
    const float4* fg4  = (const float4*)feat_lng;
    const float4* fbe4 = (const float4*)feat_lnb;

    float sg = 0.f, qg = 0.f, sf = 0.f, qf = 0.f;
    for (int i = sub; i < HIDDEN / 4; i += 8) {
      float4 g = og4[i], gb = gb4[i], f = of4[i], fb = fb4[i];
      float g0 = g.x + gb.x, g1 = g.y + gb.y, g2 = g.z + gb.z, g3 = g.w + gb.w;
      float f0 = f.x + fb.x, f1 = f.y + fb.y, f2 = f.z + fb.z, f3 = f.w + fb.w;
      sg += g0 + g1 + g2 + g3;
      qg += g0 * g0 + g1 * g1 + g2 * g2 + g3 * g3;
      sf += f0 + f1 + f2 + f3;
      qf += f0 * f0 + f1 * f1 + f2 * f2 + f3 * f3;
    }
#pragma unroll
    for (int m = 4; m >= 1; m >>= 1) {                 // reduce over 8 lanes
      sg += __shfl_xor(sg, m, 8);
      qg += __shfl_xor(qg, m, 8);
      sf += __shfl_xor(sf, m, 8);
      qf += __shfl_xor(qf, m, 8);
    }
    const float inv_d = 1.0f / (float)HIDDEN;
    float mu_g = sg * inv_d, mu_f = sf * inv_d;
    float rs_g = rsqrtf(qg * inv_d - mu_g * mu_g + 1e-5f);
    float rs_f = rsqrtf(qf * inv_d - mu_f * mu_f + 1e-5f);

    if (row0 + row < n_nodes) {
      int c = cluster[row0 + row];
      float* dst = sums + (size_t)c * HIDDEN;
      for (int i = sub; i < HIDDEN / 4; i += 8) {
        float4 g = og4[i], gb = gb4[i], f = of4[i], fb = fb4[i];
        float4 gg = gg4[i], gbe = gbe4[i], fg = fg4[i], fbe = fbe4[i];
        float v0 = gated_val(g.x + gb.x, f.x + fb.x, mu_g, rs_g, gg.x, gbe.x, mu_f, rs_f, fg.x, fbe.x);
        float v1 = gated_val(g.y + gb.y, f.y + fb.y, mu_g, rs_g, gg.y, gbe.y, mu_f, rs_f, fg.y, fbe.y);
        float v2 = gated_val(g.z + gb.z, f.z + fb.z, mu_g, rs_g, gg.z, gbe.z, mu_f, rs_f, fg.z, fbe.z);
        float v3 = gated_val(g.w + gb.w, f.w + fb.w, mu_g, rs_g, gg.w, gbe.w, mu_f, rs_f, fg.w, fbe.w);
        atomicAdd(&dst[i * 4 + 0], v0);
        atomicAdd(&dst[i * 4 + 1], v1);
        atomicAdd(&dst[i * 4 + 2], v2);
        atomicAdd(&dst[i * 4 + 3], v3);
      }
      if (sub == 0) atomicAdd(&counts[c], 1.0f);
    }
  }
}

// ---- out[n][:] = sums[cluster[n]][:] / max(count, 1)   (sums/counts in L2)
__global__ void k_gather(const float* __restrict__ sums,
                         const float* __restrict__ counts,
                         const int*   __restrict__ cluster,
                         float* __restrict__ out, int n_nodes) {
  int g = blockIdx.x * blockDim.x + threadIdx.x;
  if (g >= n_nodes * (HIDDEN / 4)) return;
  int n = g >> 7;                                      // 128 float4 per node
  int q = g & 127;
  int c = cluster[n];
  float inv = 1.0f / fmaxf(counts[c], 1.0f);
  float4 v = ((const float4*)(sums + (size_t)c * HIDDEN))[q];
  v.x *= inv; v.y *= inv; v.z *= inv; v.w *= inv;
  ((float4*)out)[g] = v;
}

extern "C" void kernel_launch(void* const* d_in, const int* in_sizes, int n_in,
                              void* d_out, int out_size, void* d_ws, size_t ws_size,
                              hipStream_t stream) {
  const float* x        = (const float*)d_in[0];
  const int*   cluster  = (const int*)d_in[1];
  // d_in[2] = batch (unused), d_in[3] = num_clusters (compile-time 2048)
  const float* gate_w   = (const float*)d_in[4];
  const float* gate_b   = (const float*)d_in[5];
  const float* gate_lng = (const float*)d_in[6];
  const float* gate_lnb = (const float*)d_in[7];
  const float* feat_w   = (const float*)d_in[8];
  const float* feat_b   = (const float*)d_in[9];
  const float* feat_lng = (const float*)d_in[10];
  const float* feat_lnb = (const float*)d_in[11];
  float* out = (float*)d_out;
  const int n_nodes = in_sizes[0] / HIDDEN;

  // workspace layout: [f16 swizzled weights 1MB][sums 4MB][counts 8KB]
  const size_t BFRAG_DWORDS = (size_t)2 * NT_TILES * KT_TILES * 32 * 8;  // 262144
  uint32_t* bfrag  = (uint32_t*)d_ws;
  float*    sums   = (float*)((char*)d_ws + BFRAG_DWORDS * 4);
  float*    counts = sums + (size_t)NCLUST * HIDDEN;

  const int nzero = NCLUST * HIDDEN + NCLUST;
  k_zero<<<(nzero + 255) / 256, 256, 0, stream>>>(sums, nzero);
  k_swizzle_w<<<(int)(BFRAG_DWORDS / 256), 256, 0, stream>>>(gate_w, feat_w, bfrag);

  const int nblocks = (n_nodes + BM - 1) / BM;         // 3125 for N=100000
  const size_t lds = 32768 + (size_t)2 * BM * HIDDEN * 4;  // 160 KB
  k_fused<<<nblocks, 256, lds, stream>>>(x, cluster, bfrag,
                                         gate_b, gate_lng, gate_lnb,
                                         feat_b, feat_lng, feat_lnb,
                                         sums, counts, n_nodes);

  const int ngather = n_nodes * (HIDDEN / 4);
  k_gather<<<(ngather + 255) / 256, 256, 0, stream>>>(sums, counts, cluster, out, n_nodes);
}
